// GatedDeltaNet_70188355551286
// MI455X (gfx1250) — matrix operand order repaired and checked by reading, not compiled, not run on hardware
//
#include <hip/hip_runtime.h>
#include <hip/hip_bf16.h>
#include <math.h>

// ---------------- CDNA5 WMMA types / helpers ----------------
typedef __bf16 bf16;
typedef __attribute__((ext_vector_type(16))) __bf16 v16bf;
typedef __attribute__((ext_vector_type(8)))  float  v8f;

#define DEV __device__ __forceinline__

// gfx1250 async global->LDS copy path (compile-safe fallback).
// Signature (probe-verified): (int4 AS1*, int4 AS3*, imm int, imm int).
#if defined(__gfx1250__) && __has_builtin(__builtin_amdgcn_global_load_async_to_lds_b128) && __has_builtin(__builtin_amdgcn_s_wait_asynccnt)
#define USE_ASYNC_COPY 1
typedef int v4i_ __attribute__((ext_vector_type(4)));
typedef __attribute__((address_space(1))) v4i_* as1_v4i;
typedef __attribute__((address_space(3))) v4i_* as3_v4i;
#else
#define USE_ASYNC_COPY 0
#endif

DEV bf16 f2bf(float f) { return (bf16)f; }   // hw v_cvt, RTNE on gfx1250
DEV float sigmoidf_(float x) { return 1.f / (1.f + __expf(-x)); }

DEV v8f wmma_bf16(v16bf a, v16bf b, v8f c) {
  return __builtin_amdgcn_wmma_f32_16x16x32_bf16(false, a, false, b, (short)0, c,
                                                 false, false);
}
// 16 contiguous bf16 (32B) -> fragment; LLVM merges into 2x ds_load_b128.
DEV v16bf ld16(const bf16* p) {
  v16bf v;
#pragma unroll
  for (int i = 0; i < 16; ++i) v[i] = p[i];
  return v;
}

// ---------------- elementwise f32 -> bf16 convert ----------------------------
__global__ __launch_bounds__(256) void f32_to_bf16_kernel(
    const float* __restrict__ src, bf16* __restrict__ dst, long long n)
{
  long long i = ((long long)blockIdx.x * 256 + threadIdx.x) * 4;
  if (i >= n) return;
  float4 v = *(const float4*)(src + i);
  union { bf16 h[4]; uint2 u; } o;
  o.h[0] = f2bf(v.x); o.h[1] = f2bf(v.y); o.h[2] = f2bf(v.z); o.h[3] = f2bf(v.w);
  *(uint2*)(dst + i) = o.u;
}

// ---------------- bf16 GEMM via WMMA, double-buffered LDS --------------------
// C[M,N](f32) = A[M,K](bf16) @ B[K,N](bf16). Block tile 128x128, K-step 32,
// 256 thr = 8 waves, each wave owns a 32x64 sub-tile (2x4 fragments).
// A tiles arrive via async DMA (ASYNCcnt), B tiles are transposed into
// sBt[n][k] so fragments are contiguous 32B runs -> ds_load_b128.
// Two LDS buffers: tile k+1 staging overlaps tile k WMMA.
__global__ __launch_bounds__(256) void gemm_bf16_wmma(
    const bf16* __restrict__ A, const bf16* __restrict__ Bm,
    float* __restrict__ C, int M, int N, int K)
{
  __shared__ __attribute__((aligned(16))) bf16 sA[2][128 * 40];   // 20 KB
  __shared__ __attribute__((aligned(16))) bf16 sBt[2][128 * 40];  // 20 KB
  const int tid = threadIdx.x, lane = tid & 31, wave = tid >> 5;
  const int wm = wave & 3, wn = wave >> 2;
  const int m0 = blockIdx.y * 128, n0 = blockIdx.x * 128;
  const int half = lane >> 4, lr = lane & 15;

  v8f acc[2][4] = {};

  auto stageA = [&](int k0, int buf) {
#pragma unroll
    for (int i = 0; i < 2; ++i) {
      int q = tid + i * 256;             // 512 chunks of 16B
      int ra = q >> 2, sa = q & 3;
#if USE_ASYNC_COPY
      __builtin_amdgcn_global_load_async_to_lds_b128(
          (as1_v4i)&A[(size_t)(m0 + ra) * K + k0 + sa * 8],
          (as3_v4i)&sA[buf][ra * 40 + sa * 8], 0, 0);
#else
      *(uint4*)&sA[buf][ra * 40 + sa * 8] =
          *(const uint4*)&A[(size_t)(m0 + ra) * K + k0 + sa * 8];
#endif
    }
  };
  auto stageB = [&](int k0, int buf) {
#pragma unroll
    for (int i = 0; i < 2; ++i) {
      int q = tid + i * 256;
      int kb_ = q >> 4, sb_ = q & 15;    // row k, 8-col segment
      union { uint4 u; bf16 h[8]; } t;
      t.u = *(const uint4*)&Bm[(size_t)(k0 + kb_) * N + n0 + sb_ * 8];
#pragma unroll
      for (int j = 0; j < 8; ++j) sBt[buf][(sb_ * 8 + j) * 40 + kb_] = t.h[j];
    }
  };

  // prologue: stage tile 0
  stageA(0, 0);
  stageB(0, 0);
#if USE_ASYNC_COPY
  __builtin_amdgcn_s_wait_asynccnt(0);
#endif
  __syncthreads();

  int buf = 0;
  for (int k0 = 0; k0 < K; k0 += 32) {
    const bool more = (k0 + 32 < K);
    // stage next tile into the other buffer while computing this one
    if (more) {
      stageA(k0 + 32, buf ^ 1);
      stageB(k0 + 32, buf ^ 1);
      if (k0 + 64 < K) {   // gfx1250 global_prefetch_b8 two tiles ahead
        __builtin_prefetch(&A[(size_t)(m0 + (tid >> 1)) * K + k0 + 64 + (tid & 1) * 16], 0, 1);
        __builtin_prefetch(&Bm[(size_t)(k0 + 64 + (tid >> 3)) * N + n0 + (tid & 7) * 16], 0, 1);
      }
    }

    v16bf af[2], bfr[4];
#pragma unroll
    for (int fm = 0; fm < 2; ++fm) {
      const bf16* pr = &sA[buf][(wm * 32 + fm * 16 + lr) * 40];
      // A-layout: K = half*16 + i, contiguous (matches B)
#pragma unroll
      for (int i = 0; i < 8; ++i) af[fm][i] = pr[half * 16 + i];
#pragma unroll
      for (int i = 0; i < 8; ++i) af[fm][8 + i] = pr[half * 16 + 8 + i];
    }
#pragma unroll
    for (int fn = 0; fn < 4; ++fn)   // B-layout: K = half*16 + i, contiguous (matches A)
      bfr[fn] = ld16(&sBt[buf][(wn * 64 + fn * 16 + lr) * 40 + half * 16]);

#pragma unroll
    for (int fm = 0; fm < 2; ++fm)
#pragma unroll
      for (int fn = 0; fn < 4; ++fn)
        acc[fm][fn] = wmma_bf16(af[fm], bfr[fn], acc[fm][fn]);

#if USE_ASYNC_COPY
    if (more) __builtin_amdgcn_s_wait_asynccnt(0);
#endif
    __syncthreads();
    buf ^= 1;
  }
  // C/D layout: VGPR r -> M = r + 8*half, N = lane&15
#pragma unroll
  for (int fm = 0; fm < 2; ++fm)
#pragma unroll
    for (int fn = 0; fn < 4; ++fn)
#pragma unroll
      for (int r = 0; r < 8; ++r) {
        int row = m0 + wm * 32 + fm * 16 + half * 8 + r;
        int col = n0 + wn * 64 + fn * 16 + lr;
        C[(size_t)row * N + col] = acc[fm][fn][r];
      }
}

// ---------------- beta / alpha gate projections ------------------------------
__global__ void proj_gates_kernel(const float* __restrict__ x,
                                  const float* __restrict__ Wb,
                                  const float* __restrict__ Wa,
                                  float* __restrict__ beta,
                                  float* __restrict__ alpha)
{
  int m = blockIdx.x;
  int n = threadIdx.x & 15;
  bool isA = threadIdx.x >= 16;
  const float* W = isA ? Wa : Wb;
  const float* xr = x + (size_t)m * 2048;
  float acc = 0.f;
  for (int k = 0; k < 2048; ++k) acc += xr[k] * W[k * 16 + n];
  (isA ? alpha : beta)[(size_t)m * 16 + n] = sigmoidf_(acc);
}

// ---------------- depthwise causal conv (KW=4) + SiLU ------------------------
__global__ __launch_bounds__(256) void conv_silu_kernel(
    const float* __restrict__ qkvz, const float* __restrict__ cw,
    float* __restrict__ qkvc, int T, long long total)
{
  long long idx = (long long)blockIdx.x * 256 + threadIdx.x;
  if (idx >= total) return;
  int c = (int)(idx % 6144);
  long long bt = idx / 6144;
  int t = (int)(bt % T);
  long long brow = bt - t;
  float acc = 0.f;
#pragma unroll
  for (int j = 0; j < 4; ++j) {
    int ts = t - 3 + j;
    if (ts >= 0) acc += cw[c * 4 + j] * qkvz[(brow + ts) * 8192 + c];
  }
  qkvc[bt * 6144 + c] = acc * sigmoidf_(acc);
}

// ---------------- per-head L2 norm of q and k (in place) ---------------------
__global__ __launch_bounds__(256) void l2norm_kernel(float* __restrict__ qk)
{
  int gid = blockIdx.x * 8 + (threadIdx.x >> 5);
  int lane = threadIdx.x & 31;
  int h = gid & 31;
  size_t bt = (size_t)(gid >> 5);
  float* p = qk + bt * 6144 + (size_t)h * 128;
  float v[4]; float ss = 0.f;
#pragma unroll
  for (int j = 0; j < 4; ++j) { v[j] = p[lane + j * 32]; ss += v[j] * v[j]; }
#pragma unroll
  for (int m = 16; m >= 1; m >>= 1) ss += __shfl_xor(ss, m);
  float inv = 1.f / fmaxf(sqrtf(ss), 1e-12f);
#pragma unroll
  for (int j = 0; j < 4; ++j) p[lane + j * 32] = v[j] * inv;
}

// ---------------- chunked gated linear-attention scan (WMMA) -----------------
// One block per (b,h). f32 state S[128x128] + transposed bf16 shadow Sb[dv][dk]
// in LDS. Chunk L=32. All WMMA operands staged so fragments are contiguous.
__global__ __launch_bounds__(256) void scan_chunk32_kernel(
    const float* __restrict__ qkv, const float* __restrict__ beta,
    const float* __restrict__ alpha, float* __restrict__ outp, int T)
{
  extern __shared__ __attribute__((aligned(16))) unsigned char smem_raw[];
  float* S   = (float*)smem_raw;                  // 128*128 f32      65536 B
  bf16*  Sb  = (bf16*)(S + 128 * 128);            // 128*136 bf16     34816 B
  bf16*  sQ  = Sb + 128 * 136;                    // 32*136 bf16       8704 B
  bf16*  sK  = sQ + 32 * 136;                     // 32*136 bf16       8704 B
  bf16*  sVt = sK + 32 * 136;                     // 128*40 bf16      10240 B
  bf16*  sPb = sVt + 128 * 40;                    // 32*40 bf16        2560 B
  float* sg  = (float*)(sPb + 32 * 40);           // g[0..31], gL at [32]

  const int b = blockIdx.x >> 4, h = blockIdx.x & 15;
  const int tid = threadIdx.x, lane = tid & 31, wave = tid >> 5;
  const int half = lane >> 4, lr = lane & 15;
  const int mrow = wave & 1, ncp = wave >> 1;

  for (int i = tid; i < 128 * 128; i += 256) S[i] = 0.f;
  for (int i = tid; i < 128 * 136; i += 256) Sb[i] = f2bf(0.f);
  __syncthreads();

  for (int c0 = 0; c0 < T; c0 += 32) {
    if (tid == 0) {
      float g = 1.f;
      for (int s = 0; s < 32; ++s) {
        g *= alpha[((size_t)b * T + c0 + s) * 16 + h];
        sg[s] = g;
      }
      sg[32] = g;
    }
    __syncthreads();
    for (int i = tid; i < 32 * 128; i += 256) {
      int s = i >> 7, d = i & 127;
      size_t row = (size_t)b * T + c0 + s;
      const float* base = qkv + row * 6144;
      float g = sg[s];
      float bt = beta[row * 16 + h];
      sQ[s * 136 + d]  = f2bf(base[(size_t)h * 128 + d] * g);
      sK[s * 136 + d]  = f2bf(base[2048 + (size_t)h * 128 + d] * (bt / g));
      sVt[d * 40 + s]  = f2bf(base[4096 + (size_t)h * 128 + d]);
    }
    __syncthreads();

    v8f oacc[2] = {};
    v8f pacc[2] = {};
#pragma unroll
    for (int kb = 0; kb < 4; ++kb) {
      v16bf aq;
      {
        const bf16* pr = &sQ[(mrow * 16 + lr) * 136 + kb * 32];
#pragma unroll
        for (int i = 0; i < 8; ++i) aq[i] = pr[half * 16 + i];
#pragma unroll
        for (int i = 0; i < 8; ++i) aq[8 + i] = pr[half * 16 + 8 + i];
      }
      // inter: out += Q~ @ S  (B from transposed bf16 shadow, contiguous)
#pragma unroll
      for (int f = 0; f < 2; ++f) {
        v16bf bs = ld16(&Sb[(ncp * 32 + f * 16 + lr) * 136 + kb * 32 + half * 16]);
        oacc[f] = wmma_bf16(aq, bs, oacc[f]);
      }
      // P = Q~ @ K~^T  (B[k=dk,n=s] = sK[s][dk], contiguous along dk)
#pragma unroll
      for (int sf = 0; sf < 2; ++sf) {
        v16bf bkT = ld16(&sK[(sf * 16 + lr) * 136 + kb * 32 + half * 16]);
        pacc[sf] = wmma_bf16(aq, bkT, pacc[sf]);
      }
    }
    if (ncp == 0) {   // write masked P as bf16, fragment-ready row-major
#pragma unroll
      for (int sf = 0; sf < 2; ++sf)
#pragma unroll
        for (int r = 0; r < 8; ++r) {
          int t = mrow * 16 + half * 8 + r;
          int s = sf * 16 + lr;
          sPb[t * 40 + s] = f2bf((s <= t) ? pacc[sf][r] : 0.f);
        }
    }
    __syncthreads();
    // intra: out += mask(P) @ V
    {
      v16bf ap;
      {
        const bf16* pr = &sPb[(mrow * 16 + lr) * 40];
#pragma unroll
        for (int i = 0; i < 8; ++i) ap[i] = pr[half * 16 + i];
#pragma unroll
        for (int i = 0; i < 8; ++i) ap[8 + i] = pr[half * 16 + 8 + i];
      }
#pragma unroll
      for (int f = 0; f < 2; ++f) {
        v16bf bv = ld16(&sVt[(ncp * 32 + f * 16 + lr) * 40 + half * 16]);
        oacc[f] = wmma_bf16(ap, bv, oacc[f]);
      }
    }
#pragma unroll
    for (int f = 0; f < 2; ++f)
#pragma unroll
      for (int r = 0; r < 8; ++r) {
        int t = mrow * 16 + half * 8 + r;
        int dv = ncp * 32 + f * 16 + lr;
        outp[((size_t)b * T + c0 + t) * 2048 + (size_t)h * 128 + dv] = oacc[f][r];
      }
    __syncthreads();
    // S = gL * (S + K~^T V); wave w owns DK rows [16w,16w+16)
    {
      float gL = sg[32];
      v16bf akT;
#pragma unroll
      for (int i = 0; i < 16; ++i) {
        int kk = half * 16 + i; // k = s
        akT[i] = sK[kk * 136 + wave * 16 + lr];
      }
#pragma unroll
      for (int fn = 0; fn < 8; ++fn) {
        v16bf bv = ld16(&sVt[(fn * 16 + lr) * 40 + half * 16]);
        v8f d = {};
        d = wmma_bf16(akT, bv, d);
#pragma unroll
        for (int r = 0; r < 8; ++r) {
          int dk = wave * 16 + half * 8 + r;
          int dv = fn * 16 + lr;
          float ns = gL * (S[dk * 128 + dv] + d[r]);
          S[dk * 128 + dv] = ns;
          Sb[dv * 136 + dk] = f2bf(ns);   // keep transposed bf16 shadow in sync
        }
      }
    }
    __syncthreads();
  }
}

// ---------------- RMS norm + sigmoid(z) gate -> bf16 activations -------------
__global__ __launch_bounds__(256) void rmsgate_kernel(
    const float* __restrict__ attn, const float* __restrict__ qkvz,
    const float* __restrict__ norm_w, bf16* __restrict__ yb)
{
  int gid = blockIdx.x * 8 + (threadIdx.x >> 5);
  int lane = threadIdx.x & 31;
  int h = gid & 15;
  size_t bt = (size_t)(gid >> 4);
  const float* o = attn + bt * 2048 + (size_t)h * 128;
  const float* z = qkvz + bt * 8192 + 6144 + (size_t)h * 128;
  float v[4]; float ss = 0.f;
#pragma unroll
  for (int j = 0; j < 4; ++j) { v[j] = o[lane + j * 32]; ss += v[j] * v[j]; }
#pragma unroll
  for (int m = 16; m >= 1; m >>= 1) ss += __shfl_xor(ss, m);
  float inv = rsqrtf(ss * (1.f / 128.f) + 1e-6f);
#pragma unroll
  for (int j = 0; j < 4; ++j) {
    int d = lane + j * 32;
    yb[bt * 2048 + (size_t)h * 128 + d] =
        f2bf(v[j] * inv * norm_w[d] * sigmoidf_(z[d]));
  }
}

// ---------------- host-side orchestration ------------------------------------
extern "C" void kernel_launch(void* const* d_in, const int* in_sizes, int n_in,
                              void* d_out, int out_size, void* d_ws, size_t ws_size,
                              hipStream_t stream) {
  (void)in_sizes; (void)n_in; (void)out_size; (void)ws_size;
  const float* x      = (const float*)d_in[0];
  // d_in[1] = positions (unused)
  const float* W_qkvz = (const float*)d_in[2];
  const float* W_b    = (const float*)d_in[3];
  const float* W_a    = (const float*)d_in[4];
  const float* conv_w = (const float*)d_in[5];
  const float* norm_w = (const float*)d_in[6];
  const float* W_out  = (const float*)d_in[7];
  float* out = (float*)d_out;

  const int T = 2048;
  const size_t BT = 8192;

  float* qkvz  = (float*)d_ws;            // BT*8192 f32
  float* qkvc  = qkvz + BT * 8192;        // BT*6144 f32
  float* beta  = qkvc + BT * 6144;        // BT*16
  float* alpha = beta + BT * 16;          // BT*16
  float* attn  = alpha + BT * 16;         // BT*2048 f32
  bf16*  xb    = (bf16*)(attn + BT * 2048);   // BT*2048 bf16
  bf16*  w1b   = xb + BT * 2048;              // 2048*8192 bf16
  bf16*  w2b   = w1b + (size_t)2048 * 8192;   // 2048*2048 bf16
  bf16*  yb    = w2b + (size_t)2048 * 2048;   // BT*2048 bf16

  dim3 blk(256);

  // 0) one-time f32 -> bf16 conversions of GEMM operands
  long long nx = (long long)BT * 2048, nw1 = 2048LL * 8192, nw2 = 2048LL * 2048;
  f32_to_bf16_kernel<<<(unsigned)((nx / 4 + 255) / 256), blk, 0, stream>>>(x, xb, nx);
  f32_to_bf16_kernel<<<(unsigned)((nw1 / 4 + 255) / 256), blk, 0, stream>>>(W_qkvz, w1b, nw1);
  f32_to_bf16_kernel<<<(unsigned)((nw2 / 4 + 255) / 256), blk, 0, stream>>>(W_out, w2b, nw2);

  // 1) qkvz = x @ W_qkvz
  gemm_bf16_wmma<<<dim3(8192 / 128, 8192 / 128), blk, 0, stream>>>(
      xb, w1b, qkvz, 8192, 8192, 2048);
  // 2) beta/alpha gates
  proj_gates_kernel<<<(unsigned)BT, 32, 0, stream>>>(x, W_b, W_a, beta, alpha);
  // 3) depthwise causal conv + SiLU
  long long convTotal = (long long)BT * 6144;
  conv_silu_kernel<<<(unsigned)((convTotal + 255) / 256), blk, 0, stream>>>(
      qkvz, conv_w, qkvc, T, convTotal);
  // 4) per-head L2 norm of q and k
  l2norm_kernel<<<(unsigned)(BT * 32 / 8), blk, 0, stream>>>(qkvc);
  // 5) chunked gated linear-attention scan
  size_t smem = 65536 + 34816 + 8704 + 8704 + 10240 + 2560 + 256; // 130816 B
  scan_chunk32_kernel<<<64, blk, smem, stream>>>(qkvc, beta, alpha, attn, T);
  // 6) RMS norm + sigmoid(z) gate -> bf16
  rmsgate_kernel<<<(unsigned)(BT * 16 / 8), blk, 0, stream>>>(attn, qkvz, norm_w, yb);
  // 7) out = gated @ W_out
  gemm_bf16_wmma<<<dim3(2048 / 128, 8192 / 128), blk, 0, stream>>>(
      yb, w2b, out, 8192, 2048, 2048);
}
